// ZeroOrderSemiCRFLayer_32160715112507
// MI455X (gfx1250) — compile-verified, matrix operand order checked by reading, not run on hardware
//
#include <hip/hip_runtime.h>
#include <stdint.h>

// ---------------------------------------------------------------------------
// Semi-CRF loss for MI455X (gfx1250, wave32).
//
//   Phase 1 (bandwidth-bound, 128 MB @ 23.3 TB/s ~ 5.5us floor):
//       row_lse[e][b][l] = logsumexp_t(transitions[b,e,l,:])
//       One wave32 per 16x32 tile; exp row sums via v_wmma_f32_16x16x32_f16
//       against an all-ones B matrix (16 row sums in one matrix op).
//       2D grid (e-waves x batches) -> no integer division in the hot kernel.
//   Phase 2 (latency-bound): order-16 logsumexp scan, ONE BATCH PER LANE
//       (64 lanes = 2 waves, one workgroup). Window lives in registers;
//       in-lane max/sum trees keep the serial chain pure-VALU. Operands are
//       staged LDS-side with double-buffered GLOBAL_LOAD_ASYNC_TO_LDS_B128 +
//       s_wait_asynccnt (CDNA5 async path), removing L2 latency from the
//       recurrence critical path.
//   Phase 3: deterministic 64-way final sum (no FP atomics).
// ---------------------------------------------------------------------------

typedef __attribute__((ext_vector_type(16))) _Float16 v16h;
typedef __attribute__((ext_vector_type(8)))  float    v8f;
typedef __attribute__((ext_vector_type(4)))  float    v4f;

#define NEG_BIG (-1.0e9f)
#define L_SEG  16
#define T_TAGS 32
#define CH      8      // scan steps staged per async chunk
#define BMAX   64      // max batches handled by the single scan workgroup
#define WPB     8      // waves per block in phase 1

typedef __attribute__((address_space(3))) float lds_float;

// One 16-byte async global->LDS copy (per-lane), tracked by ASYNCcnt.
__device__ __forceinline__ void async_copy_b128(const float* gsrc, float* lds_generic) {
  lds_float* lp = (lds_float*)lds_generic;   // VGPR holds raw LDS byte offset
  asm volatile("global_load_async_to_lds_b128 %0, %1, off"
               :: "v"(lp), "v"(gsrc)
               : "memory");
}

// ---------------------------------------------------------------------------
// Phase 1: per-(b,e) tile row-logsumexp using WMMA for the exp row sums.
// Loads are placed directly in the 16-bit A-fragment layout:
//   lane m (m<16): row m, K in {0..7,16..23};  lane m+16: K in {8..15,24..31}
// Output layout is scan-friendly: row_lse[(e*B + b)*16 + l].
// ---------------------------------------------------------------------------
__global__ __launch_bounds__(WPB * 32)
void semicrf_rowlse_wmma(const float* __restrict__ trans,
                         float* __restrict__ row_lse,
                         int S, int B) {
  const int lane = (int)(threadIdx.x & 31);
  const int e    = (int)blockIdx.x * WPB + (int)(threadIdx.x >> 5);
  const int b    = (int)blockIdx.y;
  if (e >= S) return;  // wave-uniform; EXEC stays all-ones for WMMA

  const int m  = lane & 15;   // row within tile
  const int hf = lane >> 4;   // which K half-pattern this lane covers

  const float* base = trans + ((size_t)b * S + e) * (L_SEG * T_TAGS);
  const float* p0 = base + m * T_TAGS + hf * 8;   // cols hf*8 .. hf*8+7
  const float* p1 = p0 + 16;                      // cols hf*8+16 .. hf*8+23

  float v[16];
  *(v4f*)(v + 0)  = *(const v4f*)(p0);
  *(v4f*)(v + 4)  = *(const v4f*)(p0 + 4);
  *(v4f*)(v + 8)  = *(const v4f*)(p1);
  *(v4f*)(v + 12) = *(const v4f*)(p1 + 4);

  // full-row max: 16 local values, then combine with partner lane (m <-> m+16)
  float mx = v[0];
  #pragma unroll
  for (int i = 1; i < 16; ++i) mx = fmaxf(mx, v[i]);
  mx = fmaxf(mx, __shfl_xor(mx, 16));

  // exp(tr - rowmax) packed straight into the WMMA A fragment (f16)
  v16h A;
  #pragma unroll
  for (int i = 0; i < 16; ++i) A[i] = (_Float16)__expf(v[i] - mx);

  // all-ones B matrix: layout-invariant, turns the matmul into row sums
  v16h Bones;
  #pragma unroll
  for (int i = 0; i < 16; ++i) Bones[i] = (_Float16)1.0f;

  v8f C = {0.f, 0.f, 0.f, 0.f, 0.f, 0.f, 0.f, 0.f};
  v8f D = __builtin_amdgcn_wmma_f32_16x16x32_f16(
      /*neg_a=*/false, A, /*neg_b=*/false, Bones,
      /*c_mod=*/(short)0, C, /*reuse_a=*/false, /*reuse_b=*/false);

  // D VGPR r: lanes 0..15 hold row r, lanes 16..31 hold row r+8.
  const int r = lane & 15;
  float s = D[0];
  #pragma unroll
  for (int i = 1; i < 8; ++i) if (r == i) s = D[i];

  const int srcLane = (lane < 16) ? lane : (lane - 8);
  const float rmx = __shfl(mx, srcLane);

  if (r < 8) {
    const int row = r + hf * 8;
    row_lse[((size_t)e * B + b) * L_SEG + row] = rmx + __logf(s);
  }
}

// ---------------------------------------------------------------------------
// Phase 2: scan with one batch per lane. Single workgroup of B threads.
// row_lse is [S][B][16]; chunk ci covers steps [ci*CH, ci*CH+CH) and is a
// contiguous CH*B*16*4-byte block -> double-buffered async copy into LDS.
// ---------------------------------------------------------------------------
__global__ __launch_bounds__(BMAX)
void semicrf_scan_async(const float* __restrict__ row_lse,   // [S][B][16]
                        const float* __restrict__ trans,     // [B,S,512]
                        const int* __restrict__ tags,        // [B,S]
                        const int* __restrict__ lens,        // [B]
                        float* __restrict__ partial,         // [B]
                        int S, int B) {
  __shared__ __align__(16) float lbuf[2][CH * BMAX * L_SEG];

  const int t   = (int)threadIdx.x;                 // batch id
  const int nch = (S + CH - 1) / CH;
  const size_t totalF = (size_t)S * B * L_SEG;
  const int unitsPerChunk = (CH * B * L_SEG) / 4;   // b128 units per chunk

  auto issue = [&](int ci) {
    const size_t baseF = (size_t)ci * CH * B * L_SEG;
    for (int u = t; u < unitsPerChunk; u += (int)blockDim.x) {
      const size_t fo = baseF + (size_t)u * 4;
      if (fo + 4 <= totalF)
        async_copy_b128(row_lse + fo, &lbuf[ci & 1][u * 4]);
    }
  };

  issue(0);
  if (nch > 1) issue(1);

  const int len = lens[t];
  // window: w[k] = alpha[e-k] (alpha[0]=0, alpha[j<0]=-1e9, ref padding)
  float w[L_SEG], nw[L_SEG];
  #pragma unroll
  for (int i = 0; i < L_SEG; ++i) { w[i] = NEG_BIG; nw[i] = NEG_BIG; }
  w[0] = 0.0f; nw[0] = 0.0f;
  float den_f = 0.0f, num_f = 0.0f;

  for (int ci = 0; ci < nch; ++ci) {
    // steady state: chunks ci, ci+1 outstanding (32 per-lane loads each,
    // in-order completion) -> waiting <=32 guarantees chunk ci has landed.
    if (ci + 1 < nch) asm volatile("s_wait_asynccnt 32" ::: "memory");
    else              asm volatile("s_wait_asynccnt 0"  ::: "memory");
    __syncthreads();   // all waves' copies visible before reads

    const float* lb = lbuf[ci & 1];
    const int e0 = ci * CH;

    #pragma unroll
    for (int sstep = 0; sstep < CH; ++sstep) {
      const int e = e0 + sstep;
      if (e < S) {
        float c[L_SEG];
        const float* src = lb + (sstep * B + t) * L_SEG;
        *(v4f*)(c + 0)  = *(const v4f*)(src + 0);
        *(v4f*)(c + 4)  = *(const v4f*)(src + 4);
        *(v4f*)(c + 8)  = *(const v4f*)(src + 8);
        *(v4f*)(c + 12) = *(const v4f*)(src + 12);

        float x[L_SEG];
        #pragma unroll
        for (int i = 0; i < L_SEG; ++i) x[i] = w[i] + c[i];

        // in-lane max tree (depth 4, pure VALU on the serial chain)
        float m8[8], m4[4], m2[2];
        #pragma unroll
        for (int i = 0; i < 8; ++i) m8[i] = fmaxf(x[2*i], x[2*i+1]);
        #pragma unroll
        for (int i = 0; i < 4; ++i) m4[i] = fmaxf(m8[2*i], m8[2*i+1]);
        m2[0] = fmaxf(m4[0], m4[1]); m2[1] = fmaxf(m4[2], m4[3]);
        const float mx = fmaxf(m2[0], m2[1]);

        float ex[L_SEG];
        #pragma unroll
        for (int i = 0; i < L_SEG; ++i) ex[i] = __expf(x[i] - mx);
        float s8[8], s4[4], s2[2];
        #pragma unroll
        for (int i = 0; i < 8; ++i) s8[i] = ex[2*i] + ex[2*i+1];
        #pragma unroll
        for (int i = 0; i < 4; ++i) s4[i] = s8[2*i] + s8[2*i+1];
        s2[0] = s4[0] + s4[1]; s2[1] = s4[2] + s4[3];
        const float den_new = mx + __logf(s2[0] + s2[1]);

        // numerator: general flattened (l',t') gather; idx = l'*T + t'
        const int tg = tags[(size_t)t * S + e];
        const int lp = tg >> 5;   // / T_TAGS
        const float g = trans[((size_t)t * S + e) * (L_SEG * T_TAGS) + tg];
        float a_num = nw[0];
        #pragma unroll
        for (int i = 1; i < L_SEG; ++i) if (lp == i) a_num = nw[i];
        const float num_new = a_num + g;

        if (e == len - 1) { den_f = den_new; num_f = num_new; }

        // shift window (fully unrolled -> register renaming, no scratch)
        #pragma unroll
        for (int i = L_SEG - 1; i > 0; --i) { w[i] = w[i-1]; nw[i] = nw[i-1]; }
        w[0] = den_new; nw[0] = num_new;
      }
    }

    __syncthreads();                    // everyone done reading buf[ci&1]
    if (ci + 2 < nch) issue(ci + 2);    // safe to overwrite it now
  }

  partial[t] = den_f - num_f;
}

// ---------------------------------------------------------------------------
// Phase 3: deterministic scalar reduction (no FP atomics -> reproducible).
// ---------------------------------------------------------------------------
__global__ void semicrf_reduce(const float* __restrict__ partial,
                               float* __restrict__ out, int B) {
  if (threadIdx.x == 0 && blockIdx.x == 0) {
    float s = 0.0f;
    for (int i = 0; i < B; ++i) s += partial[i];
    out[0] = s;
  }
}

extern "C" void kernel_launch(void* const* d_in, const int* in_sizes, int n_in,
                              void* d_out, int out_size, void* d_ws, size_t ws_size,
                              hipStream_t stream) {
  (void)n_in; (void)out_size; (void)ws_size;
  const float* trans = (const float*)d_in[0];  // [B,S,L,T] fp32
  const int*   tags  = (const int*)d_in[1];    // [B,S] int32
  const int*   lens  = (const int*)d_in[2];    // [B] int32
  float*       out   = (float*)d_out;          // scalar fp32

  const int B = in_sizes[2];
  const int S = in_sizes[1] / B;

  float* row_lse = (float*)d_ws;                                      // S*B*16 f32
  float* partial = (float*)((char*)d_ws +
                            (size_t)S * B * L_SEG * sizeof(float));   // B f32

  dim3 grid1((S + WPB - 1) / WPB, B);
  semicrf_rowlse_wmma<<<grid1, WPB * 32, 0, stream>>>(trans, row_lse, S, B);
  semicrf_scan_async<<<1, B, 0, stream>>>(row_lse, trans, tags, lens, partial, S, B);
  semicrf_reduce<<<1, 32, 0, stream>>>(partial, out, B);
}
